// Decoder_33492154974813
// MI455X (gfx1250) — matrix operand order repaired and checked by reading, not compiled, not run on hardware
//
#include <hip/hip_runtime.h>

// ---------------------------------------------------------------------------
// Types
// ---------------------------------------------------------------------------
typedef unsigned short bfraw;
typedef __attribute__((ext_vector_type(8)))  __bf16 v8bf;
typedef __attribute__((ext_vector_type(16))) __bf16 v16bf;
typedef __attribute__((ext_vector_type(8)))  float  v8f;
typedef __attribute__((ext_vector_type(4)))  int    v4i;
typedef __attribute__((address_space(1))) v4i as1_v4i;   // global <4 x i32>
typedef __attribute__((address_space(3))) v4i as3_v4i;   // LDS    <4 x i32>

static __device__ __forceinline__ bfraw f2bf(float f) {
  union { float f; unsigned u; } v; v.f = f;
  unsigned r = v.u + 0x7FFFu + ((v.u >> 16) & 1u);   // round-to-nearest-even
  return (bfraw)(r >> 16);
}
static __device__ __forceinline__ float bf2f(bfraw b) {
  union { unsigned u; float f; } v; v.u = ((unsigned)b) << 16;
  return v.f;
}
static __device__ __forceinline__ float sigmoidf(float x) {
  return 1.0f / (1.0f + __expf(-x));
}

// ---------------------------------------------------------------------------
// f32 -> bf16 conversion (grid-stride)
// ---------------------------------------------------------------------------
__global__ void cvt_bf16_kernel(const float* __restrict__ src,
                                bfraw* __restrict__ dst, long n) {
  long i = (long)blockIdx.x * blockDim.x + threadIdx.x;
  long stride = (long)gridDim.x * blockDim.x;
  for (; i < n; i += stride) dst[i] = f2bf(src[i]);
}

__global__ void bias_sum_kernel(const float* __restrict__ a,
                                const float* __restrict__ b,
                                float* __restrict__ o, int n) {
  int i = blockIdx.x * blockDim.x + threadIdx.x;
  if (i < n) o[i] = a[i] + b[i];
}

__global__ void init_state_kernel(const float* __restrict__ h0,
                                  const float* __restrict__ c0,
                                  float* __restrict__ h, float* __restrict__ c,
                                  bfraw* __restrict__ hbf, int n) {
  int i = blockIdx.x * blockDim.x + threadIdx.x;
  if (i < n) { h[i] = h0[i]; c[i] = c0[i]; hbf[i] = f2bf(h0[i]); }
}

// ---------------------------------------------------------------------------
// WMMA GEMM:  C[M,N] = A[M,K] @ W[N,K]^T  (+bias), bf16 in, f32 accumulate.
// grid = (tilesN/8, M/16); block = 256 (8 waves). All 8 waves of a block
// share one 16xK A tile, async-staged into LDS (ASYNCcnt path), then each
// wave streams its own 64-wide strip of W from global/L2 into 4 WMMAs/chunk.
// LDS rows padded by 8 elems (4 banks) -> conflict-free ds_load_b128.
// Requires: K <= 1024, K % 32 == 0, M % 16 == 0, N % 512 == 0 (here all hold).
// ---------------------------------------------------------------------------
#define A_STRIDE 1032   // 1024 + 8 element padding per LDS row

__global__ __launch_bounds__(256) void wmma_gemm_nt_kernel(
    const bfraw* __restrict__ A, int lda,
    const bfraw* __restrict__ W, int ldw,
    float* __restrict__ Cf, bfraw* __restrict__ Cb, long ldc,
    const float* __restrict__ bias,
    int M, int N, int K)
{
  __shared__ bfraw Atile[16 * A_STRIDE];   // ~33 KB

  const int lane = threadIdx.x & 31;
  const int wid  = threadIdx.x >> 5;
  const int m0 = blockIdx.y << 4;
  const int n0 = (blockIdx.x * 8 + wid) << 6;
  const int hs  = lane >> 4;               // half-wave select
  const int l15 = lane & 15;

  // ---- async stage A tile (16 rows x K) into LDS (ASYNCcnt path) ----
  {
    const int chunksPerRow = K >> 3;                 // 16B chunks per row
    const int totalChunks  = chunksPerRow << 4;      // 16 rows
    for (int cid = threadIdx.x; cid < totalChunks; cid += blockDim.x) {
      int row = cid / chunksPerRow;
      int cb  = (cid - row * chunksPerRow) << 4;     // byte offset within row
      const bfraw* gp = A + (long)(m0 + row) * lda + (cb >> 1);
      unsigned loff = (unsigned)(uintptr_t)Atile + (unsigned)(row * (A_STRIDE * 2) + cb);
      __builtin_amdgcn_global_load_async_to_lds_b128(
          (as1_v4i*)(uintptr_t)gp, (as3_v4i*)(uintptr_t)loff, 0, 0);
    }
    asm volatile("s_wait_asynccnt 0" ::: "memory");
    __syncthreads();
  }

  v8f z = {0.f,0.f,0.f,0.f,0.f,0.f,0.f,0.f};
  v8f acc0 = z, acc1 = z, acc2 = z, acc3 = z;

  const bfraw* Alds = Atile + l15 * A_STRIDE + (hs ? 16 : 0);
  const bfraw* W0 = W + (long)(n0 +  0 + l15) * ldw + (hs ? 16 : 0);
  const bfraw* W1 = W + (long)(n0 + 16 + l15) * ldw + (hs ? 16 : 0);
  const bfraw* W2 = W + (long)(n0 + 32 + l15) * ldw + (hs ? 16 : 0);
  const bfraw* W3 = W + (long)(n0 + 48 + l15) * ldw + (hs ? 16 : 0);

  for (int k0 = 0; k0 < K; k0 += 32) {
    // A fragment from LDS (ds_load_b128 x2, bank-conflict-free)
    v8bf lo = *reinterpret_cast<const v8bf*>(Alds + k0);
    v8bf hi = *reinterpret_cast<const v8bf*>(Alds + k0 + 8);
    v16bf af;
#pragma unroll
    for (int i = 0; i < 8; ++i) { af[i] = lo[i]; af[i + 8] = hi[i]; }

    __builtin_prefetch((const void*)(W0 + k0 + 32), 0, 1); // global_prefetch_b8

    // B fragments stream from global/L2 (weights stay L2-resident)
    v16bf b0 = *reinterpret_cast<const v16bf*>(W0 + k0);
    v16bf b1 = *reinterpret_cast<const v16bf*>(W1 + k0);
    v16bf b2 = *reinterpret_cast<const v16bf*>(W2 + k0);
    v16bf b3 = *reinterpret_cast<const v16bf*>(W3 + k0);

    acc0 = __builtin_amdgcn_wmma_f32_16x16x32_bf16(false, af, false, b0, (short)0, acc0, false, false);
    acc1 = __builtin_amdgcn_wmma_f32_16x16x32_bf16(false, af, false, b1, (short)0, acc1, false, false);
    acc2 = __builtin_amdgcn_wmma_f32_16x16x32_bf16(false, af, false, b2, (short)0, acc2, false, false);
    acc3 = __builtin_amdgcn_wmma_f32_16x16x32_bf16(false, af, false, b3, (short)0, acc3, false, false);
  }

  v8f accs[4] = {acc0, acc1, acc2, acc3};
#pragma unroll
  for (int j = 0; j < 4; ++j) {
    const int col = n0 + j * 16 + l15;
    const float bv = bias ? bias[col] : 0.0f;
#pragma unroll
    for (int r = 0; r < 8; ++r) {
      const int row = m0 + r + (hs ? 8 : 0);
      const float v = accs[j][r] + bv;
      const long idx = (long)row * ldc + col;
      if (Cf) Cf[idx] = v;
      if (Cb) Cb[idx] = f2bf(v);
    }
  }
}

// ---------------------------------------------------------------------------
// LSTM cell elementwise: gate order i,f,g,o (torch). Updates c, h, h_bf16.
// ---------------------------------------------------------------------------
__global__ void lstm_cell_kernel(const float* __restrict__ gates,
                                 float* __restrict__ c, float* __restrict__ h,
                                 bfraw* __restrict__ hbf, int Bn, int Hn) {
  int idx = blockIdx.x * blockDim.x + threadIdx.x;
  if (idx >= Bn * Hn) return;
  int b  = idx / Hn;
  int hh = idx - b * Hn;
  const float* g = gates + (long)b * 4 * Hn;
  float gi = sigmoidf(g[hh]);
  float gf = sigmoidf(g[Hn + hh]);
  float gg = tanhf(g[2 * Hn + hh]);
  float go = sigmoidf(g[3 * Hn + hh]);
  float cn = gf * c[idx] + gi * gg;
  c[idx] = cn;
  float hn = go * tanhf(cn);
  h[idx] = hn;
  hbf[idx] = f2bf(hn);
}

// ---------------------------------------------------------------------------
// Attention: one workgroup (512 threads) per batch element.
// scores[s] = q[b]·keys[b,s] -> softmax -> av[h] = sum_s attn*enc[b,s,h]
// Keys + encodings (134 MB bf16) stay resident in the 192 MB L2.
// ---------------------------------------------------------------------------
__global__ __launch_bounds__(512) void attention_kernel(
    const bfraw* __restrict__ q, const bfraw* __restrict__ keys,
    const bfraw* __restrict__ enc, bfraw* __restrict__ av,
    int Sn, int Hn)
{
  const int b = blockIdx.x;
  __shared__ float sc[512];
  __shared__ float red[32];
  const int lane = threadIdx.x & 31;
  const int wid  = threadIdx.x >> 5;
  const int nw   = blockDim.x >> 5;

  const bfraw* qb = q    + (long)b * Hn;
  const bfraw* kb = keys + (long)b * Sn * Hn;
  const bfraw* eb = enc  + (long)b * Sn * Hn;

  // ---- scores: one wave per s; v8bf per lane (512B/wave/instr) ----
  const v8bf* q8 = reinterpret_cast<const v8bf*>(qb);
  const int nch = Hn >> 3;
  for (int s = wid; s < Sn; s += nw) {
    const v8bf* k8 = reinterpret_cast<const v8bf*>(kb + (long)s * Hn);
    float acc = 0.0f;
    for (int c = lane; c < nch; c += 32) {
      v8bf kv = k8[c];
      v8bf qv = q8[c];
#pragma unroll
      for (int i = 0; i < 8; ++i) acc += (float)kv[i] * (float)qv[i];
    }
#pragma unroll
    for (int off = 16; off > 0; off >>= 1) acc += __shfl_xor(acc, off, 32);
    if (lane == 0) sc[s] = acc;
  }
  __syncthreads();

  // ---- softmax: block max ----
  float m = -3.4e38f;
  for (int s = threadIdx.x; s < Sn; s += blockDim.x) m = fmaxf(m, sc[s]);
#pragma unroll
  for (int off = 16; off > 0; off >>= 1) m = fmaxf(m, __shfl_xor(m, off, 32));
  if (lane == 0) red[wid] = m;
  __syncthreads();
  if (wid == 0) {
    float t = (lane < nw) ? red[lane] : -3.4e38f;
#pragma unroll
    for (int off = 16; off > 0; off >>= 1) t = fmaxf(t, __shfl_xor(t, off, 32));
    if (lane == 0) red[0] = t;
  }
  __syncthreads();
  m = red[0];
  __syncthreads();

  // ---- exp + block sum ----
  float lsum = 0.0f;
  for (int s = threadIdx.x; s < Sn; s += blockDim.x) {
    float e = __expf(sc[s] - m);
    sc[s] = e;
    lsum += e;
  }
#pragma unroll
  for (int off = 16; off > 0; off >>= 1) lsum += __shfl_xor(lsum, off, 32);
  if (lane == 0) red[wid] = lsum;
  __syncthreads();
  if (wid == 0) {
    float t = (lane < nw) ? red[lane] : 0.0f;
#pragma unroll
    for (int off = 16; off > 0; off >>= 1) t += __shfl_xor(t, off, 32);
    if (lane == 0) red[0] = t;
  }
  __syncthreads();
  const float inv = 1.0f / red[0];

  // ---- av: each thread owns one packed bf16 pair (2 h values) ----
  // blockDim.x == Hn/2 (512 threads, H=1024): fully coalesced 4B/thread.
  const unsigned* ep = reinterpret_cast<const unsigned*>(eb);
  const int hw = Hn >> 1;
  float a0 = 0.0f, a1 = 0.0f;
  for (int s = 0; s < Sn; ++s) {
    float w = sc[s];
    unsigned pk = ep[(long)s * hw + threadIdx.x];
    a0 += w * bf2f((bfraw)(pk & 0xFFFFu));
    a1 += w * bf2f((bfraw)(pk >> 16));
  }
  unsigned o = ((unsigned)f2bf(a1 * inv) << 16) | (unsigned)f2bf(a0 * inv);
  reinterpret_cast<unsigned*>(av)[(long)b * hw + threadIdx.x] = o;
}

// ---------------------------------------------------------------------------
// Host launcher
// ---------------------------------------------------------------------------
extern "C" void kernel_launch(void* const* d_in, const int* in_sizes, int n_in,
                              void* d_out, int out_size, void* d_ws, size_t ws_size,
                              hipStream_t stream) {
  (void)in_sizes; (void)n_in; (void)out_size; (void)ws_size;
  const int B = 64, S = 512, H = 1024, V = 4096;

  const float* enc  = (const float*)d_in[1];
  const float* h0   = (const float*)d_in[2];
  const float* c0   = (const float*)d_in[3];
  const float* Whh  = (const float*)d_in[4];
  const float* b_ih = (const float*)d_in[5];
  const float* b_hh = (const float*)d_in[6];
  const float* Wq   = (const float*)d_in[7];
  const float* Wk   = (const float*)d_in[8];
  const float* Wc   = (const float*)d_in[9];
  const float* bc   = (const float*)d_in[10];
  float* out = (float*)d_out;

  // ---- workspace carve-out (256B aligned) ----
  char* base = (char*)d_ws;
  size_t off = 0;
  auto carve = [&](size_t bytes) -> char* {
    char* r = base + off;
    off = (off + bytes + 255) & ~(size_t)255;
    return r;
  };
  bfraw* enc_bf  = (bfraw*)carve((size_t)B * S * H * 2);   // 67 MB
  bfraw* keys_bf = (bfraw*)carve((size_t)B * S * H * 2);   // 67 MB
  bfraw* whh_bf  = (bfraw*)carve((size_t)4 * H * H * 2);   // 8 MB
  bfraw* wq_bf   = (bfraw*)carve((size_t)H * H * 2);       // 2 MB
  bfraw* wk_bf   = (bfraw*)carve((size_t)H * H * 2);       // 2 MB
  bfraw* wc_bf   = (bfraw*)carve((size_t)V * H * 2);       // 8 MB
  float* bias4h  = (float*)carve((size_t)4 * H * 4);
  float* h_f32   = (float*)carve((size_t)B * H * 4);
  float* c_f32   = (float*)carve((size_t)B * H * 4);
  bfraw* h_bf    = (bfraw*)carve((size_t)B * H * 2);
  float* gates   = (float*)carve((size_t)B * 4 * H * 4);
  bfraw* q_bf    = (bfraw*)carve((size_t)B * H * 2);
  bfraw* av_bf   = (bfraw*)carve((size_t)B * H * 2);

  auto gemm = [&](const bfraw* A, int lda, const bfraw* W, int ldw,
                  float* Cf, bfraw* Cb, long ldc, const float* bias,
                  int M, int N, int K) {
    dim3 grid((N / 64) / 8, M / 16);
    wmma_gemm_nt_kernel<<<grid, 256, 0, stream>>>(A, lda, W, ldw, Cf, Cb,
                                                  ldc, bias, M, N, K);
  };

  // ---- one-time prep ----
  cvt_bf16_kernel<<<8192, 256, 0, stream>>>(enc, enc_bf, (long)B * S * H);
  cvt_bf16_kernel<<<2048, 256, 0, stream>>>(Whh, whh_bf, (long)4 * H * H);
  cvt_bf16_kernel<<<1024, 256, 0, stream>>>(Wq, wq_bf, (long)H * H);
  cvt_bf16_kernel<<<1024, 256, 0, stream>>>(Wk, wk_bf, (long)H * H);
  cvt_bf16_kernel<<<2048, 256, 0, stream>>>(Wc, wc_bf, (long)V * H);
  bias_sum_kernel<<<(4 * H + 255) / 256, 256, 0, stream>>>(b_ih, b_hh, bias4h, 4 * H);
  init_state_kernel<<<(B * H + 255) / 256, 256, 0, stream>>>(h0, c0, h_f32, c_f32,
                                                             h_bf, B * H);

  // keys = encodings @ Wk^T   ([B*S, H] x [H, H]^T -> bf16)
  gemm(enc_bf, H, wk_bf, H, nullptr, keys_bf, H, nullptr, B * S, H, H);

  // ---- recurrent decode ----
  for (int t = 0; t < S; ++t) {
    // gates = h @ W_hh^T + (b_ih + b_hh)
    gemm(h_bf, H, whh_bf, H, gates, nullptr, 4 * H, bias4h, B, 4 * H, H);
    // LSTM cell update -> c, h (f32 + bf16)
    lstm_cell_kernel<<<(B * H + 255) / 256, 256, 0, stream>>>(gates, c_f32, h_f32,
                                                              h_bf, B, H);
    // q = h @ Wq^T
    gemm(h_bf, H, wq_bf, H, nullptr, q_bf, H, nullptr, B, H, H);
    // attention -> av
    attention_kernel<<<B, 512, 0, stream>>>(q_bf, keys_bf, enc_bf, av_bf, S, H);
    // logits = av @ Wc^T + bc, written directly to out[:, t, :]
    gemm(av_bf, H, wc_bf, H, out + (long)t * V, nullptr, (long)S * V, bc, B, V, H);
  }
}
